// SE_79508434583933
// MI455X (gfx1250) — matrix operand order repaired and checked by reading, not compiled, not run on hardware
//
#include <hip/hip_runtime.h>

// ---------------------------------------------------------------------------
// Problem constants (from the reference)
// ---------------------------------------------------------------------------
#define NB   16      // batch
#define CCH  2048    // C
#define ICH  1024    // IC
#define HWSZ 1024    // H*W (== IC)

typedef __bf16 bf16;
typedef __attribute__((ext_vector_type(2)))  bf16  bf16x2;
typedef __attribute__((ext_vector_type(16))) bf16  v16bf;
typedef __attribute__((ext_vector_type(8)))  float v8f;

// ---------------------------------------------------------------------------
// bf16 helpers — native converts (v_cvt_pk_bf16_f32 on gfx1250)
// ---------------------------------------------------------------------------
static __device__ __forceinline__ bf16 f2bf(float f) { return (bf16)f; }

// pack two floats into one dword of two bf16 (single v_cvt_pk_bf16_f32)
static __device__ __forceinline__ unsigned pk2(float x, float y) {
    bf16x2 v = { (bf16)x, (bf16)y };
    return __builtin_bit_cast(unsigned, v);
}

static __device__ __forceinline__ void store_out(float* p, float v) { *p = v; }
static __device__ __forceinline__ void store_out(bf16*  p, float v) { *p = f2bf(v); }

// 8-element loader: reads 8 consecutive source elements, returns packed bf16[8]
struct Pack8 { union { uint4 q; bf16 h[8]; }; };

static __device__ __forceinline__ Pack8 load8(const float* p) {
    const float4* f4 = (const float4*)p;
    float4 a = f4[0], b = f4[1];
    Pack8 r;
    r.q.x = pk2(a.x, a.y);
    r.q.y = pk2(a.z, a.w);
    r.q.z = pk2(b.x, b.y);
    r.q.w = pk2(b.z, b.w);
    return r;
}
static __device__ __forceinline__ Pack8 load8(const bf16* p) {
    Pack8 r;
    r.q = *(const uint4*)p;
    return r;
}

union FragU { v16bf v; uint4 q[2]; };

// ---------------------------------------------------------------------------
// Tiled batched GEMM:  out[M,N] = A[M,K] * B[K,N] (+ bias[m])
//   - A, B may be fp32 (converted to bf16 while staging into LDS) or bf16
//   - out may be fp32 or bf16; optionally stored transposed (out[n*M + m])
//   - grid = (N/128, M/128, batch); block = 256 threads = 8 wave32 waves
//   - block tile 128(M) x 128(N) x 32(K); each wave owns a 32x64 tile:
//     8x v_wmma_f32_16x16x32_bf16 per K-step, fragments via ds_load_b128
// ---------------------------------------------------------------------------
template<typename AT, typename BT, typename OT, bool TRANS_OUT, bool HAS_BIAS>
__global__ __launch_bounds__(256)
void gemm_bf16_wmma(const AT* __restrict__ A, const BT* __restrict__ Bm,
                    OT* __restrict__ out, const float* __restrict__ bias,
                    int M, int N, int K,
                    long long sA, long long sB, long long sO)
{
    // A tile: row-major [m][k], padded row stride 40 elems (80 B, 16B-aligned)
    // B tile: stored TRANSPOSED [n][k] so fragment reads are contiguous
    __shared__ bf16 ldsA[128][40];
    __shared__ bf16 ldsBt[128][40];

    const int tid  = threadIdx.x;
    const int lane = tid & 31;
    const int wave = tid >> 5;          // 0..7
    const int wm   = wave >> 1;         // 0..3  (M direction, 32 rows each)
    const int wn   = wave & 1;          // 0..1  (N direction, 64 cols each)
    const int half = lane >> 4;         // 0/1
    const int lr   = lane & 15;

    const int blockM = blockIdx.y * 128;
    const int blockN = blockIdx.x * 128;

    const AT* Ab = A  + (long long)blockIdx.z * sA;
    const BT* Bb = Bm + (long long)blockIdx.z * sB;
    OT*       Ob = out + (long long)blockIdx.z * sO;

    // staging coordinates (each thread handles 8 consecutive elements/chunk)
    const int ar = tid >> 2;            // 0..63   (A rows, 2 chunks of 64)
    const int ac = (tid & 3) * 8;       // 0,8,16,24
    const int bk = tid >> 3;            // 0..31   (B row = k)
    const int bn = (tid & 7) * 8;       // 0..56   (2 chunks of 64 in n)

    v8f acc[2][4] = {{{}, {}, {}, {}}, {{}, {}, {}, {}}};

    for (int k0 = 0; k0 < K; k0 += 32) {
        // ---- stage A tile (128 x 32): packed 16-byte LDS stores ----
        #pragma unroll
        for (int q = 0; q < 2; ++q) {
            int r = ar + q * 64;
            Pack8 p = load8(&Ab[(size_t)(blockM + r) * K + (k0 + ac)]);
            *(uint4*)&ldsA[r][ac] = p.q;
        }
        // ---- stage B tile (32 x 128), transposed into [n][k] ----
        #pragma unroll
        for (int q = 0; q < 2; ++q) {
            int n0 = bn + q * 64;
            Pack8 p = load8(&Bb[(size_t)(k0 + bk) * N + (blockN + n0)]);
            #pragma unroll
            for (int j = 0; j < 8; ++j) ldsBt[n0 + j][bk] = p.h[j];
        }
        // prefetch next K tile while we compute on this one
        if (k0 + 32 < K) {
            __builtin_prefetch(&Ab[(size_t)(blockM + ar) * K + (k0 + 32 + ac)], 0, 3);
            __builtin_prefetch(&Bb[(size_t)(k0 + 32 + bk) * N + (blockN + bn)], 0, 3);
        }
        __syncthreads();

        // ---- fragments: all contiguous -> ds_load_b128 pairs ----
        FragU a[2], b[4];
        #pragma unroll
        for (int s = 0; s < 2; ++s) {
            const int r = wm * 32 + s * 16 + lr;
            a[s].q[0] = *(const uint4*)&ldsA[r][half * 8];
            a[s].q[1] = *(const uint4*)&ldsA[r][half * 8 + 16];
        }
        #pragma unroll
        for (int t = 0; t < 4; ++t) {
            const int n = wn * 64 + t * 16 + lr;
            b[t].q[0] = *(const uint4*)&ldsBt[n][half * 8];
            b[t].q[1] = *(const uint4*)&ldsBt[n][half * 8 + 16];
        }

        #pragma unroll
        for (int s = 0; s < 2; ++s)
            #pragma unroll
            for (int t = 0; t < 4; ++t)
                acc[s][t] = __builtin_amdgcn_wmma_f32_16x16x32_bf16(
                    false, a[s].v, false, b[t].v, (short)0, acc[s][t], false, false);

        __syncthreads();
    }

    // ---- epilogue: C/D layout — VGPR v: M = v + 8*half, N = lane%16 ----
    #pragma unroll
    for (int s = 0; s < 2; ++s) {
        #pragma unroll
        for (int t = 0; t < 4; ++t) {
            const int nG = blockN + wn * 64 + t * 16 + lr;
            #pragma unroll
            for (int v = 0; v < 8; ++v) {
                const int mG = blockM + wm * 32 + s * 16 + half * 8 + v;
                float val = acc[s][t][v];
                if (HAS_BIAS) val += bias[mG];
                if (TRANS_OUT) store_out(&Ob[(size_t)nG * M + mG], val);
                else           store_out(&Ob[(size_t)mG * N + nG], val);
            }
        }
    }
}

// ---------------------------------------------------------------------------
// Row softmax: one 256-thread block per row of length ncols; bf16 output
// ---------------------------------------------------------------------------
__global__ __launch_bounds__(256)
void softmax_rows_kernel(const float* __restrict__ f, bf16* __restrict__ out, int ncols)
{
    __shared__ float red[256];
    const int tid = threadIdx.x;
    const size_t row = blockIdx.x;
    const float* fr = f + row * (size_t)ncols;

    float mx = -3.402823466e38f;
    for (int i = tid; i < ncols; i += 256) mx = fmaxf(mx, fr[i]);
    red[tid] = mx; __syncthreads();
    for (int s = 128; s > 0; s >>= 1) {
        if (tid < s) red[tid] = fmaxf(red[tid], red[tid + s]);
        __syncthreads();
    }
    const float m = red[0];
    __syncthreads();

    float sum = 0.0f;
    for (int i = tid; i < ncols; i += 256) sum += __expf(fr[i] - m);
    red[tid] = sum; __syncthreads();
    for (int s = 128; s > 0; s >>= 1) {
        if (tid < s) red[tid] += red[tid + s];
        __syncthreads();
    }
    const float inv = 1.0f / red[0];

    bf16* orow = out + row * (size_t)ncols;
    for (int i = tid; i < ncols; i += 256) orow[i] = f2bf(__expf(fr[i] - m) * inv);
}

// ---------------------------------------------------------------------------
// BatchNorm statistics: one block per channel c; reduce over (b, hw)
// ---------------------------------------------------------------------------
__global__ __launch_bounds__(256)
void bn_stats_kernel(const float* __restrict__ wy,
                     float* __restrict__ mean, float* __restrict__ invstd)
{
    __shared__ float rs[256], rq[256];
    const int c = blockIdx.x;
    const int tid = threadIdx.x;
    float s = 0.0f, q = 0.0f;
    for (int i = tid; i < NB * HWSZ; i += 256) {
        int b  = i >> 10;
        int hw = i & (HWSZ - 1);
        float v = wy[((size_t)b * CCH + c) * HWSZ + hw];
        s += v; q += v * v;
    }
    rs[tid] = s; rq[tid] = q; __syncthreads();
    for (int st = 128; st > 0; st >>= 1) {
        if (tid < st) { rs[tid] += rs[tid + st]; rq[tid] += rq[tid + st]; }
        __syncthreads();
    }
    if (tid == 0) {
        const float n = (float)(NB * HWSZ);
        float mn  = rs[0] / n;
        float var = rq[0] / n - mn * mn;
        mean[c]   = mn;
        invstd[c] = rsqrtf(var + 1e-5f);
    }
}

// ---------------------------------------------------------------------------
// Final: out = (wy - mean)*invstd*gamma + beta + x_this
// ---------------------------------------------------------------------------
__global__ __launch_bounds__(256)
void finalize_kernel(const float* __restrict__ wy, const float* __restrict__ x_this,
                     const float* __restrict__ mean, const float* __restrict__ invstd,
                     const float* __restrict__ gamma, const float* __restrict__ beta,
                     float* __restrict__ out, size_t n)
{
    size_t i = (size_t)blockIdx.x * blockDim.x + threadIdx.x;
    if (i >= n) return;
    int c = (int)((i >> 10) & (CCH - 1));
    out[i] = (wy[i] - mean[c]) * invstd[c] * gamma[c] + beta[c] + x_this[i];
}

// ---------------------------------------------------------------------------
// Workspace layout (bytes)
// ---------------------------------------------------------------------------
static const size_t OFF_GX    = 0;                         // bf16 (B,IC,HW)   33.5 MB
static const size_t OFF_THETA = OFF_GX    + 33554432;      // bf16 (B,IC,HW)   33.5 MB (reused for y)
static const size_t OFF_PHIT  = OFF_THETA + 33554432;      // bf16 (B,HW,IC)   33.5 MB (reused for softmax(f))
static const size_t OFF_F     = OFF_PHIT  + 33554432;      // f32  (B,HW,HW)   67 MB
static const size_t OFF_WY    = OFF_F     + 67108864;      // f32  (B,C,HW)    134 MB
static const size_t OFF_STAT  = OFF_WY    + 134217728;     // f32  mean[2048] + invstd[2048]

extern "C" void kernel_launch(void* const* d_in, const int* in_sizes, int n_in,
                              void* d_out, int out_size, void* d_ws, size_t ws_size,
                              hipStream_t stream) {
    const float* x_this  = (const float*)d_in[0];
    const float* x_other = (const float*)d_in[1];
    const float* g_w     = (const float*)d_in[2];
    const float* g_b     = (const float*)d_in[3];
    const float* theta_w = (const float*)d_in[4];
    const float* theta_b = (const float*)d_in[5];
    const float* phi_w   = (const float*)d_in[6];
    const float* phi_b   = (const float*)d_in[7];
    const float* wz_w    = (const float*)d_in[8];
    const float* wz_b    = (const float*)d_in[9];
    const float* gamma   = (const float*)d_in[10];
    const float* beta    = (const float*)d_in[11];
    float* out = (float*)d_out;

    char* ws = (char*)d_ws;
    bf16*  g_x    = (bf16*)(ws + OFF_GX);
    bf16*  theta  = (bf16*)(ws + OFF_THETA);
    bf16*  phiT   = (bf16*)(ws + OFF_PHIT);
    float* f      = (float*)(ws + OFF_F);
    bf16*  fbf    = (bf16*)(ws + OFF_PHIT);    // reuse phiT
    bf16*  y      = (bf16*)(ws + OFF_THETA);   // reuse theta
    float* wy     = (float*)(ws + OFF_WY);
    float* mean   = (float*)(ws + OFF_STAT);
    float* invstd = mean + CCH;

    const long long sX  = (long long)CCH * HWSZ;   // per-batch x stride
    const long long sM  = (long long)ICH * HWSZ;   // per-batch (IC,HW) stride
    const long long sWY = (long long)CCH * HWSZ;

    dim3 blk(256);

    // 1) g_x = g_w (IC x C) * x_this (C x HW) + g_b        -> bf16 (B,IC,HW)
    gemm_bf16_wmma<float, float, bf16, false, true>
        <<<dim3(HWSZ / 128, ICH / 128, NB), blk, 0, stream>>>(
        g_w, x_this, g_x, g_b, ICH, HWSZ, CCH, 0, sX, sM);

    // 2) theta_x = theta_w * x_this + theta_b              -> bf16 (B,IC,HW)
    gemm_bf16_wmma<float, float, bf16, false, true>
        <<<dim3(HWSZ / 128, ICH / 128, NB), blk, 0, stream>>>(
        theta_w, x_this, theta, theta_b, ICH, HWSZ, CCH, 0, sX, sM);

    // 3) phi_x = phi_w * x_other + phi_b, stored TRANSPOSED -> bf16 (B,HW,IC)
    gemm_bf16_wmma<float, float, bf16, true, true>
        <<<dim3(HWSZ / 128, ICH / 128, NB), blk, 0, stream>>>(
        phi_w, x_other, phiT, phi_b, ICH, HWSZ, CCH, 0, sX, sM);

    // 4) f = phiT (HW x IC) * theta (IC x HW)              -> f32 (B,HW,HW)
    gemm_bf16_wmma<bf16, bf16, float, false, false>
        <<<dim3(HWSZ / 128, HWSZ / 128, NB), blk, 0, stream>>>(
        phiT, theta, f, nullptr, HWSZ, HWSZ, ICH, sM, sM, sM);

    // 5) softmax over last axis                            -> bf16 (B,HW,HW)
    softmax_rows_kernel<<<NB * HWSZ, blk, 0, stream>>>(f, fbf, HWSZ);

    // 6) y = softmax(f) (HW x HW) * g_x (IC x HW)          -> bf16 (B,HW,HW)
    gemm_bf16_wmma<bf16, bf16, bf16, false, false>
        <<<dim3(HWSZ / 128, HWSZ / 128, NB), blk, 0, stream>>>(
        fbf, g_x, y, nullptr, HWSZ, HWSZ, HWSZ, sM, sM, sM);

    // 7) wy = wz_w (C x IC) * y (IC x HW) + wz_b           -> f32 (B,C,HW)
    gemm_bf16_wmma<float, bf16, float, false, true>
        <<<dim3(HWSZ / 128, CCH / 128, NB), blk, 0, stream>>>(
        wz_w, y, wy, wz_b, CCH, HWSZ, ICH, 0, sM, sWY);

    // 8) BatchNorm statistics (per channel over B,HW)
    bn_stats_kernel<<<CCH, blk, 0, stream>>>(wy, mean, invstd);

    // 9) normalize + affine + residual
    const size_t n = (size_t)NB * CCH * HWSZ;
    finalize_kernel<<<(unsigned)((n + 255) / 256), blk, 0, stream>>>(
        wy, x_this, mean, invstd, gamma, beta, out, n);
}
